// TFN_15607911154160
// MI455X (gfx1250) — compile-verified, matrix operand order checked by reading, not run on hardware
//
#include <hip/hip_runtime.h>

// ---------------------------------------------------------------------------
// TFN forward for MI455X (gfx1250, wave32, WMMA).
// Dominant cost: fusion @ Wp1.T  (Wp1 = 558 MB fp32 -> HBM bound at 23.3TB/s,
// ~24us floor). Strategy: stream Wp1 exactly once (disjoint K partitions
// across 16x65 blocks), stage it into double-buffered LDS with gfx1250
// async-to-LDS DMA (ASYNCcnt), synthesize the rank-1 fusion operand in
// registers with packed bf16 multiplies, accumulate f32 WMMA, and combine
// partials with global f32 atomics.
// ---------------------------------------------------------------------------

#define EPSV  1e-5f
#define BB    256
#define SEQ   50
#define TIN   300
#define TINP  320        // TIN padded to K-chunk multiple
#define HID   128
#define G4    512        // 4*HID
#define AIN   74
#define VIN   35
#define SUBH  64         // audio/video hidden
#define TO_   128        // text output width
#define PF    256
#define FUSN  545025LL   // 65*65*129
#define AVN   4225       // 65*65
#define TPAD  160        // 129 padded to 5 K-chunks of 32

typedef __attribute__((ext_vector_type(16))) __bf16 v16bf;
typedef __attribute__((ext_vector_type(2)))  __bf16 bf16x2;
typedef __attribute__((ext_vector_type(8)))  float  v8f;

#if __has_builtin(__builtin_amdgcn_global_load_async_to_lds_b32) && \
    __has_builtin(__builtin_amdgcn_s_wait_asynccnt)
#define HAVE_ASYNC_LDS 1
typedef __attribute__((address_space(1))) int as1i;
typedef __attribute__((address_space(3))) int as3i;
#else
#define HAVE_ASYNC_LDS 0
#endif

// K position of element-pair j (elements 2j,2j+1) for the 16-bit 16x32 A/B
// fragment layout (lanes 0-15: K 0-7 / 16-23; lanes 16-31: K 8-15 / 24-31).
__device__ __forceinline__ int kpair(int j, int hf) {
  return (j < 4) ? (hf * 8 + j * 2) : (16 + hf * 8 + (j - 4) * 2);
}

// Load a 16-wide bf16 fragment; rowbase points at (row, kchunk*32), rows are
// selected per-lane by the caller (lane & 15). 4-byte packed loads.
__device__ __forceinline__ v16bf load_bf16_frag(const __bf16* rowbase, int hf) {
  union { v16bf v; unsigned int u[8]; } r;
  const unsigned int* p = (const unsigned int*)rowbase;
#pragma unroll
  for (int j = 0; j < 8; ++j) r.u[j] = p[kpair(j, hf) >> 1];
  return r.v;
}

// Build a bf16 fragment from an f32 row (LDS-staged Wp1 tile).
__device__ __forceinline__ v16bf cvt_frag_f32(const float* rowbase, int hf) {
  union { v16bf v; bf16x2 h[8]; } r;
#pragma unroll
  for (int j = 0; j < 8; ++j) {
    int k = kpair(j, hf);
    float2 f = *(const float2*)(rowbase + k);   // k even -> 8B aligned
    bf16x2 c; c[0] = (__bf16)f.x; c[1] = (__bf16)f.y;
    r.h[j] = c;
  }
  return r.v;
}

// Synthesize fusion A-fragment: frag[k] = av_scalar * T[b, tbase+k].
// Packed bf16 multiply -> v_pk_mul_bf16 (VOP3P).
__device__ __forceinline__ v16bf make_fusion_frag(const __bf16* trow, bf16x2 avp, int hf) {
  union { v16bf v; bf16x2 h[8]; } r;
  const bf16x2* p = (const bf16x2*)trow;
#pragma unroll
  for (int j = 0; j < 8; ++j) r.h[j] = p[kpair(j, hf) >> 1] * avp;
  return r.v;
}

__device__ __forceinline__ float sigf(float x) { return 1.f / (1.f + __expf(-x)); }

// ------------------------- small VALU kernels ------------------------------

__global__ void bn_stats_kernel(const float* __restrict__ x, int F,
                                float* __restrict__ mean, float* __restrict__ var) {
  int f = blockIdx.x * blockDim.x + threadIdx.x;
  if (f >= F) return;
  float s = 0.f, ss = 0.f;
  for (int b = 0; b < BB; ++b) { float v = x[(size_t)b * F + f]; s += v; ss += v * v; }
  float m = s * (1.f / BB);
  mean[f] = m;
  var[f]  = ss * (1.f / BB) - m * m;
}

__global__ void fc_kernel(const float* __restrict__ in, const float* __restrict__ W,
                          const float* __restrict__ bias, float* __restrict__ out,
                          int B, int Fin, int Fout, int dorelu,
                          const float* __restrict__ mean, const float* __restrict__ var,
                          const float* __restrict__ gamma, const float* __restrict__ beta) {
  int idx = blockIdx.x * blockDim.x + threadIdx.x;
  if (idx >= B * Fout) return;
  int b = idx / Fout, j = idx % Fout;
  const float* xr = in + (size_t)b * Fin;
  const float* wr = W + (size_t)j * Fin;
  float acc = bias[j];
  for (int f = 0; f < Fin; ++f) {
    float v = xr[f];
    if (mean) v = (v - mean[f]) * rsqrtf(var[f] + EPSV) * gamma[f] + beta[f];
    acc += v * wr[f];
  }
  out[idx] = dorelu ? fmaxf(acc, 0.f) : acc;
}

__global__ void cvt_pad_kernel(const float* __restrict__ in, __bf16* __restrict__ out,
                               int rows, int kin, int kout) {
  int idx = blockIdx.x * blockDim.x + threadIdx.x;
  if (idx >= rows * kout) return;
  int r = idx / kout, k = idx % kout;
  out[idx] = (k < kin) ? (__bf16)in[(size_t)r * kin + k] : (__bf16)0.f;
}

__global__ void zero_u32_kernel(unsigned int* __restrict__ p, int n) {
  int i = blockIdx.x * blockDim.x + threadIdx.x;
  if (i < n) p[i] = 0u;
}

__global__ void prep_av_kernel(const float* __restrict__ ah, const float* __restrict__ vh,
                               __bf16* __restrict__ avbf) {
  int idx = blockIdx.x * blockDim.x + threadIdx.x;
  if (idx >= BB * AVN) return;
  int b = idx / AVN, av = idx % AVN;
  int a = av / 65, v = av % 65;
  float fa = (a == 0) ? 1.f : ah[(size_t)b * SUBH + a - 1];
  float fv = (v == 0) ? 1.f : vh[(size_t)b * SUBH + v - 1];
  avbf[idx] = (__bf16)(fa * fv);
}

__global__ void prep_t_kernel(const float* __restrict__ th, __bf16* __restrict__ tbf) {
  int idx = blockIdx.x * blockDim.x + threadIdx.x;
  if (idx >= BB * TPAD) return;
  int b = idx / TPAD, t = idx % TPAD;
  float v = (t == 0) ? 1.f : ((t < 129) ? th[(size_t)b * TO_ + t - 1] : 0.f);
  tbf[idx] = (__bf16)v;
}

__global__ void bias_relu_kernel(float* __restrict__ y, const float* __restrict__ bias,
                                 int B, int F) {
  int idx = blockIdx.x * blockDim.x + threadIdx.x;
  if (idx >= B * F) return;
  y[idx] = fmaxf(y[idx] + bias[idx % F], 0.f);
}

// ----------------- pre-activation GEMM: x @ Wih.T + (bih+bhh) --------------
// M = B*SEQ = 12800 rows (row r = b*SEQ + s), N = 512, K = 320 (padded).
__global__ void __launch_bounds__(256)
pre_gemm_kernel(const __bf16* __restrict__ xbf, const __bf16* __restrict__ wihbf,
                const float* __restrict__ bih, const float* __restrict__ bhh,
                float* __restrict__ pre) {
  const int lane = threadIdx.x & 31;
  const int hf = lane >> 4, ln = lane & 15;
  const int waveg = blockIdx.x * 8 + (threadIdx.x >> 5);
  const int mbase = (waveg >> 5) * 16;   // 800 M-tiles
  const int nbase = (waveg & 31) * 16;   // 32  N-tiles
  const float bv = bih[nbase + ln] + bhh[nbase + ln];
  v8f acc = {bv, bv, bv, bv, bv, bv, bv, bv};
  const __bf16* arow = xbf   + (size_t)(mbase + ln) * TINP;
  const __bf16* brow = wihbf + (size_t)(nbase + ln) * TINP;
#pragma unroll
  for (int kc = 0; kc < TINP / 32; ++kc) {
    v16bf a = load_bf16_frag(arow + kc * 32, hf);
    v16bf b = load_bf16_frag(brow + kc * 32, hf);
    acc = __builtin_amdgcn_wmma_f32_16x16x32_bf16(false, a, false, b, (short)0,
                                                  acc, false, false);
  }
#pragma unroll
  for (int r = 0; r < 8; ++r) {
    int m = r + 8 * hf;
    pre[(size_t)(mbase + m) * G4 + nbase + ln] = acc[r];
  }
}

// ----------------- one LSTM step: z = pre[s] + h @ Whh.T, gates ------------
// 128 waves: 16 batch tiles x 8 hidden-col tiles; each wave computes the
// i/f/g/o tiles for its (b,h) block with WMMA, then the cell update entirely
// in registers using the C/D layout (m = r + 8*(lane/16), n = lane%16).
__global__ void __launch_bounds__(256)
lstm_step_kernel(const float* __restrict__ pre, const __bf16* __restrict__ hin,
                 __bf16* __restrict__ hout, const __bf16* __restrict__ whhbf,
                 float* __restrict__ cbuf, float* __restrict__ hlast, int s) {
  const int lane = threadIdx.x & 31;
  const int hf = lane >> 4, ln = lane & 15;
  const int wid = blockIdx.x * 8 + (threadIdx.x >> 5);  // 0..127
  const int mbase = (wid & 15) * 16;
  const int hbase = (wid >> 4) * 16;
  v8f ai, af, ag, ao;
#pragma unroll
  for (int r = 0; r < 8; ++r) {
    int m = r + 8 * hf;
    size_t ro = ((size_t)(mbase + m) * SEQ + s) * G4;
    ai[r] = pre[ro + 0 * HID + hbase + ln];
    af[r] = pre[ro + 1 * HID + hbase + ln];
    ag[r] = pre[ro + 2 * HID + hbase + ln];
    ao[r] = pre[ro + 3 * HID + hbase + ln];
  }
  const __bf16* arow = hin + (size_t)(mbase + ln) * HID;
#pragma unroll
  for (int kc = 0; kc < HID / 32; ++kc) {
    v16bf a = load_bf16_frag(arow + kc * 32, hf);
    v16bf b0 = load_bf16_frag(whhbf + (size_t)(0 * HID + hbase + ln) * HID + kc * 32, hf);
    v16bf b1 = load_bf16_frag(whhbf + (size_t)(1 * HID + hbase + ln) * HID + kc * 32, hf);
    v16bf b2 = load_bf16_frag(whhbf + (size_t)(2 * HID + hbase + ln) * HID + kc * 32, hf);
    v16bf b3 = load_bf16_frag(whhbf + (size_t)(3 * HID + hbase + ln) * HID + kc * 32, hf);
    ai = __builtin_amdgcn_wmma_f32_16x16x32_bf16(false, a, false, b0, (short)0, ai, false, false);
    af = __builtin_amdgcn_wmma_f32_16x16x32_bf16(false, a, false, b1, (short)0, af, false, false);
    ag = __builtin_amdgcn_wmma_f32_16x16x32_bf16(false, a, false, b2, (short)0, ag, false, false);
    ao = __builtin_amdgcn_wmma_f32_16x16x32_bf16(false, a, false, b3, (short)0, ao, false, false);
  }
#pragma unroll
  for (int r = 0; r < 8; ++r) {
    int m = r + 8 * hf;
    size_t ci = (size_t)(mbase + m) * HID + hbase + ln;
    float i = sigf(ai[r]), f = sigf(af[r]), g = tanhf(ag[r]), o = sigf(ao[r]);
    float c = f * cbuf[ci] + i * g;
    float h = o * tanhf(c);
    cbuf[ci]  = c;
    hout[ci]  = (__bf16)h;
    hlast[ci] = h;     // last step's write survives
  }
}

// ----------------- dominant kernel: y1 += fusion @ Wp1.T -------------------
// Grid: (16 P-chunks) x (65 a-slices). Each block: 8 waves x 2 M-tiles cover
// all 256 batch rows, so its disjoint Wp1 K-slice is streamed exactly once.
// Wp1 chunks (16p x 32k fp32, 2KB) are DMA'd into double-buffered LDS with
// GLOBAL_LOAD_ASYNC_TO_LDS_B32 (ASYNCcnt), one barrier per chunk. The fusion
// A-operand is synthesized in registers with v_pk_mul_bf16 from AV (constant
// per chunk, since t is padded 129->160 so a 32-wide K-chunk never crosses an
// (a,v) boundary) and the padded T factor.
__global__ void __launch_bounds__(256)
fusion_wp1_kernel(const __bf16* __restrict__ avbf, const __bf16* __restrict__ tbf,
                  const float* __restrict__ wp1, float* __restrict__ y1) {
  __shared__ float wtile[2][16][32];
  const int tid  = threadIdx.x;
  const int lane = tid & 31;
  const int wv   = tid >> 5;       // wave 0..7
  const int hf   = lane >> 4;
  const int ln   = lane & 15;
  const int pbase = blockIdx.x * 16;
  const int a0    = blockIdx.y;    // a index 0..64 (disjoint K partition)
  const int mb0 = wv * 32, mb1 = mb0 + 16;
  // staging role: thread owns Wp1[pbase+sp][sk..sk+1] of the 16x32 chunk
  const int sp = tid >> 4;
  const int sk = (tid & 15) * 2;

  v8f acc0 = {}; v8f acc1 = {};
  const int NC = 65 * 5;           // (v, t-chunk) pairs in this a-slice

  // stage chunk c into LDS buffer (c&1)
  auto stage = [&](int c) {
    const int vv = c / 5, tc = c % 5;
    const long kcol = (long)(a0 * 65 + vv) * 129;
    float* dst = &wtile[c & 1][sp][sk];
    const float* src = wp1 + (long)(pbase + sp) * FUSN + kcol + tc * 32 + sk;
    if (tc < 4) {
#if HAVE_ASYNC_LDS
      __builtin_amdgcn_global_load_async_to_lds_b32((as1i*)(src),     (as3i*)(dst),     0, 0);
      __builtin_amdgcn_global_load_async_to_lds_b32((as1i*)(src + 1), (as3i*)(dst + 1), 0, 0);
#else
      dst[0] = src[0];
      dst[1] = src[1];
      __builtin_prefetch(src + 32, 0, 1);   // global_prefetch_b8
#endif
    } else {                        // ragged tail: only t==128 is real
      dst[0] = (sk == 0) ? src[0] : 0.f;
      dst[1] = 0.f;
    }
  };

  stage(0);
  for (int c = 0; c < NC; ++c) {
#if HAVE_ASYNC_LDS
    __builtin_amdgcn_s_wait_asynccnt(0);    // chunk c DMA landed (this wave)
#endif
    __syncthreads();                        // all waves: buf(c) ready, buf(c-1) free
    if (c + 1 < NC) stage(c + 1);           // overlap next DMA with compute
    const int av = a0 * 65 + c / 5;
    const int tb = (c % 5) * 32;
    __bf16 avb0 = avbf[(size_t)(mb0 + ln) * AVN + av];
    __bf16 avb1 = avbf[(size_t)(mb1 + ln) * AVN + av];
    bf16x2 avp0; avp0[0] = avb0; avp0[1] = avb0;
    bf16x2 avp1; avp1[0] = avb1; avp1[1] = avb1;
    v16bf bfr = cvt_frag_f32(&wtile[c & 1][ln][0], hf);
    v16bf af0 = make_fusion_frag(&tbf[(size_t)(mb0 + ln) * TPAD + tb], avp0, hf);
    v16bf af1 = make_fusion_frag(&tbf[(size_t)(mb1 + ln) * TPAD + tb], avp1, hf);
    acc0 = __builtin_amdgcn_wmma_f32_16x16x32_bf16(false, af0, false, bfr, (short)0, acc0, false, false);
    acc1 = __builtin_amdgcn_wmma_f32_16x16x32_bf16(false, af1, false, bfr, (short)0, acc1, false, false);
  }
#pragma unroll
  for (int r = 0; r < 8; ++r) {
    int m = r + 8 * hf;
    atomicAdd(&y1[(size_t)(mb0 + m) * PF + pbase + ln], acc0[r]);
    atomicAdd(&y1[(size_t)(mb1 + m) * PF + pbase + ln], acc1[r]);
  }
}

// ---------------------------------------------------------------------------

extern "C" void kernel_launch(void* const* d_in, const int* in_sizes, int n_in,
                              void* d_out, int out_size, void* d_ws, size_t ws_size,
                              hipStream_t stream) {
  const float* audio_x = (const float*)d_in[0];
  const float* video_x = (const float*)d_in[1];
  const float* text_x  = (const float*)d_in[2];
  const float* ga  = (const float*)d_in[3];  const float* ba_ = (const float*)d_in[4];
  const float* Wa1 = (const float*)d_in[5];  const float* ba1 = (const float*)d_in[6];
  const float* Wa2 = (const float*)d_in[7];  const float* ba2 = (const float*)d_in[8];
  const float* Wa3 = (const float*)d_in[9];  const float* ba3 = (const float*)d_in[10];
  const float* gv  = (const float*)d_in[11]; const float* bv_ = (const float*)d_in[12];
  const float* Wv1 = (const float*)d_in[13]; const float* bv1 = (const float*)d_in[14];
  const float* Wv2 = (const float*)d_in[15]; const float* bv2 = (const float*)d_in[16];
  const float* Wv3 = (const float*)d_in[17]; const float* bv3 = (const float*)d_in[18];
  const float* Wih = (const float*)d_in[19]; const float* Whh = (const float*)d_in[20];
  const float* bih = (const float*)d_in[21]; const float* bhh = (const float*)d_in[22];
  const float* Wt  = (const float*)d_in[23]; const float* bt  = (const float*)d_in[24];
  const float* Wp1 = (const float*)d_in[25]; const float* bp1 = (const float*)d_in[26];
  const float* Wp2 = (const float*)d_in[27]; const float* bp2 = (const float*)d_in[28];
  const float* Wp3 = (const float*)d_in[29]; const float* bp3 = (const float*)d_in[30];

  char* ws = (char*)d_ws;
  size_t off = 0;
  auto alloc = [&](size_t bytes) {
    size_t o = off; off += (bytes + 255) & ~(size_t)255; return (void*)(ws + o);
  };
  float*  meanA = (float*)alloc(AIN * 4);
  float*  varA  = (float*)alloc(AIN * 4);
  float*  meanV = (float*)alloc(VIN * 4);
  float*  varV  = (float*)alloc(VIN * 4);
  float*  aw1 = (float*)alloc((size_t)BB * SUBH * 4);
  float*  aw2 = (float*)alloc((size_t)BB * SUBH * 4);
  float*  aw3 = (float*)alloc((size_t)BB * SUBH * 4);
  float*  vw1 = (float*)alloc((size_t)BB * SUBH * 4);
  float*  vw2 = (float*)alloc((size_t)BB * SUBH * 4);
  float*  vw3 = (float*)alloc((size_t)BB * SUBH * 4);
  __bf16* xbf    = (__bf16*)alloc((size_t)BB * SEQ * TINP * 2);   // 8.2 MB
  __bf16* wihbf  = (__bf16*)alloc((size_t)G4 * TINP * 2);
  __bf16* whhbf  = (__bf16*)alloc((size_t)G4 * HID * 2);
  float*  pre    = (float*)alloc((size_t)BB * SEQ * G4 * 4);      // 26.2 MB
  __bf16* hbfA   = (__bf16*)alloc((size_t)BB * HID * 2);
  __bf16* hbfB   = (__bf16*)alloc((size_t)BB * HID * 2);
  float*  cbuf   = (float*)alloc((size_t)BB * HID * 4);
  float*  hlast  = (float*)alloc((size_t)BB * HID * 4);
  float*  texth  = (float*)alloc((size_t)BB * TO_ * 4);
  __bf16* avbf   = (__bf16*)alloc((size_t)BB * AVN * 2);          // 2.2 MB
  __bf16* tbf    = (__bf16*)alloc((size_t)BB * TPAD * 2);
  float*  y1     = (float*)alloc((size_t)BB * PF * 4);
  float*  y2     = (float*)alloc((size_t)BB * PF * 4);
  (void)ws_size; (void)in_sizes; (void)n_in; (void)out_size;

  const int T = 256;
  auto blocks = [](long n) { return (unsigned)((n + 255) / 256); };

  // --- audio / video subnets (tiny; VALU) ---
  bn_stats_kernel<<<1, 128, 0, stream>>>(audio_x, AIN, meanA, varA);
  bn_stats_kernel<<<1, 128, 0, stream>>>(video_x, VIN, meanV, varV);
  fc_kernel<<<blocks(BB * SUBH), T, 0, stream>>>(audio_x, Wa1, ba1, aw1, BB, AIN, SUBH, 1, meanA, varA, ga, ba_);
  fc_kernel<<<blocks(BB * SUBH), T, 0, stream>>>(aw1, Wa2, ba2, aw2, BB, SUBH, SUBH, 1, nullptr, nullptr, nullptr, nullptr);
  fc_kernel<<<blocks(BB * SUBH), T, 0, stream>>>(aw2, Wa3, ba3, aw3, BB, SUBH, SUBH, 1, nullptr, nullptr, nullptr, nullptr);
  fc_kernel<<<blocks(BB * SUBH), T, 0, stream>>>(video_x, Wv1, bv1, vw1, BB, VIN, SUBH, 1, meanV, varV, gv, bv_);
  fc_kernel<<<blocks(BB * SUBH), T, 0, stream>>>(vw1, Wv2, bv2, vw2, BB, SUBH, SUBH, 1, nullptr, nullptr, nullptr, nullptr);
  fc_kernel<<<blocks(BB * SUBH), T, 0, stream>>>(vw2, Wv3, bv3, vw3, BB, SUBH, SUBH, 1, nullptr, nullptr, nullptr, nullptr);

  // --- LSTM: bf16 conversions, hoisted input GEMM, 50 sequential steps ---
  cvt_pad_kernel<<<blocks((long)BB * SEQ * TINP), T, 0, stream>>>(text_x, xbf, BB * SEQ, TIN, TINP);
  cvt_pad_kernel<<<blocks((long)G4 * TINP), T, 0, stream>>>(Wih, wihbf, G4, TIN, TINP);
  cvt_pad_kernel<<<blocks((long)G4 * HID), T, 0, stream>>>(Whh, whhbf, G4, HID, HID);
  pre_gemm_kernel<<<3200, T, 0, stream>>>(xbf, wihbf, bih, bhh, pre);
  zero_u32_kernel<<<blocks(BB * HID / 2), T, 0, stream>>>((unsigned int*)hbfA, BB * HID / 2);
  zero_u32_kernel<<<blocks(BB * HID), T, 0, stream>>>((unsigned int*)cbuf, BB * HID);
  for (int s = 0; s < SEQ; ++s) {
    const __bf16* hin = (s & 1) ? hbfB : hbfA;
    __bf16* hout      = (s & 1) ? hbfA : hbfB;
    lstm_step_kernel<<<16, T, 0, stream>>>(pre, hin, hout, whhbf, cbuf, hlast, s);
  }
  fc_kernel<<<blocks(BB * TO_), T, 0, stream>>>(hlast, Wt, bt, texth, BB, HID, TO_, 0, nullptr, nullptr, nullptr, nullptr);

  // --- fusion factors + dominant GEMM (stream Wp1 once) ---
  prep_av_kernel<<<blocks((long)BB * AVN), T, 0, stream>>>(aw3, vw3, avbf);
  prep_t_kernel<<<blocks((long)BB * TPAD), T, 0, stream>>>(texth, tbf);
  zero_u32_kernel<<<blocks(BB * PF), T, 0, stream>>>((unsigned int*)y1, BB * PF);
  fusion_wp1_kernel<<<dim3(16, 65), T, 0, stream>>>(avbf, tbf, Wp1, y1);
  bias_relu_kernel<<<blocks(BB * PF), T, 0, stream>>>(y1, bp1, BB, PF);

  // --- post-fusion head ---
  fc_kernel<<<blocks(BB * PF), T, 0, stream>>>(y1, Wp2, bp2, y2, BB, PF, PF, 1, nullptr, nullptr, nullptr, nullptr);
  fc_kernel<<<blocks(BB * 1), T, 0, stream>>>(y2, Wp3, bp3, (float*)d_out, BB, PF, 1, 0, nullptr, nullptr, nullptr, nullptr);
}
